// GPT_16131897164200
// MI455X (gfx1250) — matrix operand order repaired and checked by reading, not compiled, not run on hardware
//
#include <hip/hip_runtime.h>
#include <stdint.h>

// Problem constants (reference: B=8,T=512,D=768,H=12,L=6,V=8192, HD=64)
#define Bb 8
#define Tt 512
#define Dd 768
#define Hh 12
#define HDd 64
#define Ll 6
#define Vv 8192
#define BT (Bb * Tt)   // 4096
#define D4 (4 * Dd)    // 3072

typedef __attribute__((ext_vector_type(16))) __bf16 v16bf;
typedef __attribute__((ext_vector_type(8)))  float  v8f;

union FragB16 {
  v16bf v;
  unsigned short s[16];
  uint4 q[2];
};

__device__ __forceinline__ unsigned short f2bf(float f) {
  unsigned int u = __float_as_uint(f);
  u = (u + 0x7FFFu + ((u >> 16) & 1u)) >> 16;   // round-to-nearest-even
  return (unsigned short)u;
}

__device__ __forceinline__ v8f wmma_bf16(const FragB16 &a, const FragB16 &b, v8f c) {
  // 8 args: (neg_a, A, neg_b, B, c_mod, C, reuse_a, reuse_b)
  return __builtin_amdgcn_wmma_f32_16x16x32_bf16(false, a.v, false, b.v,
                                                 (short)0, c, false, false);
}

// ---------------------------------------------------------------------------
// fp32 [K,N] -> bf16 transposed [N,K], tiled through LDS for coalescing.
// blockDim = (32,8); grid = (N/32, K/32, numLayers). K,N multiples of 32.
// ---------------------------------------------------------------------------
__global__ __launch_bounds__(256) void convT_kernel(const float* __restrict__ src,
                                                    unsigned short* __restrict__ dst,
                                                    int K, int N) {
  __shared__ unsigned short tile[32][33];
  size_t mat = (size_t)blockIdx.z * K * N;
  const float* s = src + mat;
  unsigned short* d = dst + mat;
  int k0 = blockIdx.y * 32;
  int n0 = blockIdx.x * 32;
#pragma unroll
  for (int j = 0; j < 4; ++j) {
    int r = threadIdx.y + j * 8;
    tile[r][threadIdx.x] = f2bf(s[(size_t)(k0 + r) * N + n0 + threadIdx.x]);
  }
  __syncthreads();
#pragma unroll
  for (int j = 0; j < 4; ++j) {
    int r = threadIdx.y + j * 8;
    d[(size_t)(n0 + r) * K + k0 + threadIdx.x] = tile[threadIdx.x][r];
  }
}

// ---------------------------------------------------------------------------
// Embedding: x[b,t,:] = tok_emb[idx[b,t],:] + pos_emb[t,:]   (fp32)
// ---------------------------------------------------------------------------
__global__ void embed_kernel(const int* __restrict__ idx,
                             const float* __restrict__ tok,
                             const float* __restrict__ pos,
                             float* __restrict__ x) {
  int i = blockIdx.x * blockDim.x + threadIdx.x;
  if (i >= BT * Dd) return;
  int d  = i % Dd;
  int bt = i / Dd;
  int t  = bt % Tt;
  x[i] = tok[(size_t)idx[bt] * Dd + d] + pos[(size_t)t * Dd + d];
}

// ---------------------------------------------------------------------------
// LayerNorm (per row of length Dn), output bf16. One 256-thread block per row.
// ---------------------------------------------------------------------------
__global__ __launch_bounds__(256) void ln_kernel(const float* __restrict__ x,
                                                 const float* __restrict__ g,
                                                 const float* __restrict__ beta,
                                                 unsigned short* __restrict__ out,
                                                 int Dn) {
  __shared__ float red0[8];
  __shared__ float red1[8];
  int row = blockIdx.x;
  const float* xr = x + (size_t)row * Dn;
  float s = 0.f, s2 = 0.f;
  for (int i = threadIdx.x; i < Dn; i += 256) {
    float v = xr[i];
    s += v; s2 += v * v;
  }
  for (int m = 1; m < 32; m <<= 1) {
    s  += __shfl_xor(s,  m, 32);
    s2 += __shfl_xor(s2, m, 32);
  }
  int wid = threadIdx.x >> 5;
  if ((threadIdx.x & 31) == 0) { red0[wid] = s; red1[wid] = s2; }
  __syncthreads();
  s = 0.f; s2 = 0.f;
  for (int w = 0; w < 8; ++w) { s += red0[w]; s2 += red1[w]; }
  float mu  = s / (float)Dn;
  float var = s2 / (float)Dn - mu * mu;
  float rs  = rsqrtf(var + 1e-5f);
  for (int i = threadIdx.x; i < Dn; i += 256) {
    float v = (xr[i] - mu) * rs * g[i] + beta[i];
    out[(size_t)row * Dn + i] = f2bf(v);
  }
}

// ---------------------------------------------------------------------------
// bf16 WMMA GEMM with TRANSPOSED weights: C[M,N] = A[M,K] @ Bt[N,K]^T
//   mode 0: bf16 out (no bias)                          (Q,K projections)
//   mode 1: f32 out = acc + bias[n] + res[m,n]          (Wo proj, MLP2)
//   mode 2: bf16 out = relu(acc + bias[n])              (MLP1)
//   mode 3: f32 out = acc + bias[n]                     (LM head)
//   mode 4: bf16 out TRANSPOSED: outB[n*M + m]          (V projection)
// Block = 256 threads = 8 waves; each wave -> 32 rows x 64 cols,
// software-pipelined K loop (register double buffering).
// Requires M%256==0, N%64==0, K%64==0.
// ---------------------------------------------------------------------------
__global__ __launch_bounds__(256) void gemm_bf16_kernel(
    const unsigned short* __restrict__ A, const unsigned short* __restrict__ Bt,
    const float* __restrict__ bias, const float* __restrict__ res,
    float* __restrict__ outF, unsigned short* __restrict__ outB,
    int M, int N, int K, int mode) {
  int lane = threadIdx.x & 31;
  int wid  = threadIdx.x >> 5;
  int l16  = lane & 15;
  int half = lane >> 4;
  int rowBase = blockIdx.y * 256 + wid * 32;
  int colBase = blockIdx.x * 64;

  v8f acc[2][4];
#pragma unroll
  for (int rt = 0; rt < 2; ++rt)
#pragma unroll
    for (int j = 0; j < 4; ++j)
      acc[rt][j] = (v8f){0.f,0.f,0.f,0.f,0.f,0.f,0.f,0.f};

  // A: lane-half h holds K = k0 + h*8 + [0..7] and k0 + 16 + h*8 + [0..7]
  const unsigned short* arow0 = A + (size_t)(rowBase + l16) * K + half * 8;
  const unsigned short* arow1 = arow0 + (size_t)16 * K;
  // B (transposed): lane n, half h -> column colBase+j4*16+n, K = k0 + h*8 + [0..7] and k0 + 16 + h*8 + [0..7]
  const unsigned short* bcol  = Bt + (size_t)(colBase + l16) * K + half * 8;

  auto loadA = [&](FragB16* a, int k0) {
    a[0].q[0] = *(const uint4*)(arow0 + k0);
    a[0].q[1] = *(const uint4*)(arow0 + k0 + 16);
    a[1].q[0] = *(const uint4*)(arow1 + k0);
    a[1].q[1] = *(const uint4*)(arow1 + k0 + 16);
  };
  auto loadB = [&](FragB16* b, int k0) {
#pragma unroll
    for (int j4 = 0; j4 < 4; ++j4) {
      const unsigned short* bp = bcol + (size_t)(j4 * 16) * K + k0;
      b[j4].q[0] = *(const uint4*)(bp);
      b[j4].q[1] = *(const uint4*)(bp + 16);
    }
  };
  auto compute = [&](FragB16* a, FragB16* b) {
#pragma unroll
    for (int rt = 0; rt < 2; ++rt)
#pragma unroll
      for (int j4 = 0; j4 < 4; ++j4)
        acc[rt][j4] = wmma_bf16(a[rt], b[j4], acc[rt][j4]);
  };

  FragB16 a0[2], b0[4], a1[2], b1[4];
  loadA(a0, 0);
  loadB(b0, 0);
  int k0 = 0;
  for (; k0 + 64 < K; k0 += 64) {
    loadA(a1, k0 + 32);
    loadB(b1, k0 + 32);
    compute(a0, b0);
    loadA(a0, k0 + 64);
    loadB(b0, k0 + 64);
    compute(a1, b1);
  }
  loadA(a1, k0 + 32);
  loadB(b1, k0 + 32);
  compute(a0, b0);
  compute(a1, b1);

  // Epilogue: row = rowBase + rt*16 + r + 8*half, col = colBase + j4*16 + l16
#pragma unroll
  for (int rt = 0; rt < 2; ++rt) {
#pragma unroll
    for (int j4 = 0; j4 < 4; ++j4) {
      int col = colBase + j4 * 16 + l16;
#pragma unroll
      for (int r = 0; r < 8; ++r) {
        int row = rowBase + rt * 16 + r + 8 * half;
        float v = acc[rt][j4][r];
        if (mode == 0) {
          outB[(size_t)row * N + col] = f2bf(v);
        } else if (mode == 1) {
          size_t o = (size_t)row * N + col;
          outF[o] = v + bias[col] + res[o];
        } else if (mode == 2) {
          float t = v + bias[col];
          outB[(size_t)row * N + col] = f2bf(t > 0.f ? t : 0.f);
        } else if (mode == 3) {
          outF[(size_t)row * N + col] = v + bias[col];
        } else {  // mode 4: transposed bf16 store
          outB[(size_t)col * M + row] = f2bf(v);
        }
      }
    }
  }
}

// ---------------------------------------------------------------------------
// Fused causal attention, flash-style. One wave per (b, h, 16-query tile).
// Q,K stored bf16 [B*T, D] (head h = cols h*64..h*64+63).
// V stored bf16 TRANSPOSED: Vt[D, B*T]  (row d = h*64+hd, col = b*T + t).
// ---------------------------------------------------------------------------
__global__ __launch_bounds__(32) void attn_kernel(
    const unsigned short* __restrict__ Q, const unsigned short* __restrict__ Kb,
    const unsigned short* __restrict__ Vt, unsigned short* __restrict__ O) {
  __shared__ __align__(16) unsigned short ldsP[16 * 32];
  const float scale = 0.125f;   // HD^-0.5 = 1/8
  const float NEG   = -3.0e38f;

  int lane = threadIdx.x;
  int l16  = lane & 15;
  int half = lane >> 4;
  int q0 = blockIdx.x * 16;
  int h  = blockIdx.y;
  int b  = blockIdx.z;

  // Q A-fragments for K-dim (head dim) 0..31 and 32..63
  FragB16 aq0, aq1;
  {
    const unsigned short* qp = Q + (size_t)(b * Tt + q0 + l16) * Dd + h * HDd;
    aq0.q[0] = *(const uint4*)(qp + half * 8);
    aq0.q[1] = *(const uint4*)(qp + 16 + half * 8);
    aq1.q[0] = *(const uint4*)(qp + 32 + half * 8);
    aq1.q[1] = *(const uint4*)(qp + 48 + half * 8);
  }

  v8f oacc[4];
#pragma unroll
  for (int j = 0; j < 4; ++j) oacc[j] = (v8f){0.f,0.f,0.f,0.f,0.f,0.f,0.f,0.f};
  float mrow[8], lrow[8];
#pragma unroll
  for (int r = 0; r < 8; ++r) { mrow[r] = NEG; lrow[r] = 0.f; }

  for (int s0 = 0; s0 <= q0 + 15; s0 += 32) {
    // ---- scores S[16q x 32s] = Q @ K^T, two 16x16 column tiles ----
    v8f sc[2];
#pragma unroll
    for (int ct = 0; ct < 2; ++ct) {
      FragB16 bk0, bk1;
      const unsigned short* kp =
          Kb + (size_t)(b * Tt + s0 + ct * 16 + l16) * Dd + h * HDd;
      bk0.q[0] = *(const uint4*)(kp + half * 8);
      bk0.q[1] = *(const uint4*)(kp + half * 8 + 16);
      bk1.q[0] = *(const uint4*)(kp + 32 + half * 8);
      bk1.q[1] = *(const uint4*)(kp + 32 + half * 8 + 16);
      v8f z = (v8f){0.f,0.f,0.f,0.f,0.f,0.f,0.f,0.f};
      z = wmma_bf16(aq0, bk0, z);
      z = wmma_bf16(aq1, bk1, z);
      sc[ct] = z;
    }

    // ---- online softmax update (rows live across 16 lanes of each half) ----
    float p0[8], p1[8];
#pragma unroll
    for (int r = 0; r < 8; ++r) {
      int qi = q0 + r + 8 * half;
      float v0 = sc[0][r] * scale;
      float v1 = sc[1][r] * scale;
      if (s0 + l16 > qi)       v0 = NEG;
      if (s0 + 16 + l16 > qi)  v1 = NEG;
      float tm = fmaxf(v0, v1);
      tm = fmaxf(tm, __shfl_xor(tm, 1, 16));
      tm = fmaxf(tm, __shfl_xor(tm, 2, 16));
      tm = fmaxf(tm, __shfl_xor(tm, 4, 16));
      tm = fmaxf(tm, __shfl_xor(tm, 8, 16));
      float mn  = fmaxf(mrow[r], tm);
      float fac = __expf(mrow[r] - mn);
      float e0  = __expf(v0 - mn);
      float e1  = __expf(v1 - mn);
      float ps  = e0 + e1;
      ps += __shfl_xor(ps, 1, 16);
      ps += __shfl_xor(ps, 2, 16);
      ps += __shfl_xor(ps, 4, 16);
      ps += __shfl_xor(ps, 8, 16);
      lrow[r] = lrow[r] * fac + ps;
      mrow[r] = mn;
#pragma unroll
      for (int j = 0; j < 4; ++j) oacc[j][r] *= fac;
      p0[r] = e0; p1[r] = e1;
    }

    // ---- transpose P (C layout) -> A fragment via LDS ----
    __syncthreads();
#pragma unroll
    for (int r = 0; r < 8; ++r) {
      ldsP[(r + 8 * half) * 32 + l16]      = f2bf(p0[r]);
      ldsP[(r + 8 * half) * 32 + 16 + l16] = f2bf(p1[r]);
    }
    __syncthreads();
    FragB16 aP;
    aP.q[0] = *(const uint4*)&ldsP[l16 * 32 + half * 8];
    aP.q[1] = *(const uint4*)&ldsP[l16 * 32 + 16 + half * 8];

    // ---- O += P @ V  (K-dim = 32 keys); Vt rows are contiguous in t ----
#pragma unroll
    for (int j4 = 0; j4 < 4; ++j4) {
      FragB16 bv;
      const unsigned short* vp =
          Vt + (size_t)(h * HDd + j4 * 16 + l16) * BT + b * Tt + s0 + half * 8;
      bv.q[0] = *(const uint4*)(vp);
      bv.q[1] = *(const uint4*)(vp + 16);
      oacc[j4] = wmma_bf16(aP, bv, oacc[j4]);
    }
  }

  // ---- normalize and store bf16 ----
#pragma unroll
  for (int r = 0; r < 8; ++r) {
    float inv = 1.0f / lrow[r];
    int row = q0 + r + 8 * half;
#pragma unroll
    for (int j4 = 0; j4 < 4; ++j4) {
      O[(size_t)(b * Tt + row) * Dd + h * HDd + j4 * 16 + l16] =
          f2bf(oacc[j4][r] * inv);
    }
  }
}

// ---------------------------------------------------------------------------
// Host-side launcher
// ---------------------------------------------------------------------------
extern "C" void kernel_launch(void* const* d_in, const int* in_sizes, int n_in,
                              void* d_out, int out_size, void* d_ws, size_t ws_size,
                              hipStream_t stream) {
  (void)in_sizes; (void)n_in; (void)out_size; (void)ws_size;

  const int*   idx   = (const int*)  d_in[0];
  const float* tok   = (const float*)d_in[1];
  const float* pos   = (const float*)d_in[2];
  const float* Wq    = (const float*)d_in[3];
  const float* Wk    = (const float*)d_in[4];
  const float* Wv    = (const float*)d_in[5];
  const float* Wo    = (const float*)d_in[6];
  const float* bo    = (const float*)d_in[7];
  const float* W1    = (const float*)d_in[8];
  const float* b1    = (const float*)d_in[9];
  const float* W2    = (const float*)d_in[10];
  const float* b2    = (const float*)d_in[11];
  const float* ln1g  = (const float*)d_in[12];
  const float* ln1b  = (const float*)d_in[13];
  const float* ln2g  = (const float*)d_in[14];
  const float* ln2b  = (const float*)d_in[15];
  const float* lnfg  = (const float*)d_in[16];
  const float* lnfb  = (const float*)d_in[17];
  const float* Wlm   = (const float*)d_in[18];
  const float* blm   = (const float*)d_in[19];
  float* logits = (float*)d_out;

  char* cur = (char*)d_ws;
  auto take = [&](size_t bytes) -> void* {
    void* r = (void*)cur;
    cur += (bytes + 255) & ~(size_t)255;
    return r;
  };
  float*          xf   = (float*)         take((size_t)BT * Dd * 4);
  unsigned short* hbf  = (unsigned short*)take((size_t)BT * Dd * 2);
  unsigned short* qbf  = (unsigned short*)take((size_t)BT * Dd * 2);
  unsigned short* kbf  = (unsigned short*)take((size_t)BT * Dd * 2);
  unsigned short* vtb  = (unsigned short*)take((size_t)BT * Dd * 2);  // transposed V
  unsigned short* aobf = (unsigned short*)take((size_t)BT * Dd * 2);
  unsigned short* m1bf = (unsigned short*)take((size_t)BT * D4 * 2);
  unsigned short* wqb  = (unsigned short*)take((size_t)Ll * Dd * Dd * 2);
  unsigned short* wkb  = (unsigned short*)take((size_t)Ll * Dd * Dd * 2);
  unsigned short* wvb  = (unsigned short*)take((size_t)Ll * Dd * Dd * 2);
  unsigned short* wob  = (unsigned short*)take((size_t)Ll * Dd * Dd * 2);
  unsigned short* w1b  = (unsigned short*)take((size_t)Ll * Dd * D4 * 2);
  unsigned short* w2b  = (unsigned short*)take((size_t)Ll * D4 * Dd * 2);
  unsigned short* wlmb = (unsigned short*)take((size_t)Dd * Vv * 2);

  // Weight conversions fp32 [K,N] -> bf16 transposed [N,K]
  dim3 tb(32, 8);
  convT_kernel<<<dim3(Dd / 32, Dd / 32, Ll), tb, 0, stream>>>(Wq,  wqb,  Dd, Dd);
  convT_kernel<<<dim3(Dd / 32, Dd / 32, Ll), tb, 0, stream>>>(Wk,  wkb,  Dd, Dd);
  convT_kernel<<<dim3(Dd / 32, Dd / 32, Ll), tb, 0, stream>>>(Wv,  wvb,  Dd, Dd);
  convT_kernel<<<dim3(Dd / 32, Dd / 32, Ll), tb, 0, stream>>>(Wo,  wob,  Dd, Dd);
  convT_kernel<<<dim3(D4 / 32, Dd / 32, Ll), tb, 0, stream>>>(W1,  w1b,  Dd, D4);
  convT_kernel<<<dim3(Dd / 32, D4 / 32, Ll), tb, 0, stream>>>(W2,  w2b,  D4, Dd);
  convT_kernel<<<dim3(Vv / 32, Dd / 32, 1),  tb, 0, stream>>>(Wlm, wlmb, Dd, Vv);

  // Embedding
  embed_kernel<<<(BT * Dd + 255) / 256, 256, 0, stream>>>(idx, tok, pos, xf);

  dim3 gD (Dd / 64,  BT / 256);   // (12, 16)
  dim3 gD4(D4 / 64,  BT / 256);   // (48, 16)
  dim3 gV (Vv / 64,  BT / 256);   // (128, 16)

  for (int l = 0; l < Ll; ++l) {
    size_t wOff  = (size_t)l * Dd * Dd;
    size_t w1Off = (size_t)l * Dd * D4;

    // h = LN1(x)
    ln_kernel<<<BT, 256, 0, stream>>>(xf, ln1g + l * Dd, ln1b + l * Dd, hbf, Dd);
    // q,k = h @ W{q,k};  v = h @ Wv stored transposed
    gemm_bf16_kernel<<<gD, 256, 0, stream>>>(hbf, wqb + wOff, nullptr, nullptr,
                                             nullptr, qbf, BT, Dd, Dd, 0);
    gemm_bf16_kernel<<<gD, 256, 0, stream>>>(hbf, wkb + wOff, nullptr, nullptr,
                                             nullptr, kbf, BT, Dd, Dd, 0);
    gemm_bf16_kernel<<<gD, 256, 0, stream>>>(hbf, wvb + wOff, nullptr, nullptr,
                                             nullptr, vtb, BT, Dd, Dd, 4);
    // attention
    attn_kernel<<<dim3(Tt / 16, Hh, Bb), 32, 0, stream>>>(qbf, kbf, vtb, aobf);
    // x = x + attn @ Wo + bo
    gemm_bf16_kernel<<<gD, 256, 0, stream>>>(aobf, wob + wOff, bo + l * Dd, xf,
                                             xf, nullptr, BT, Dd, Dd, 1);
    // h2 = LN2(x)
    ln_kernel<<<BT, 256, 0, stream>>>(xf, ln2g + l * Dd, ln2b + l * Dd, hbf, Dd);
    // m1 = relu(h2 @ W1 + b1)
    gemm_bf16_kernel<<<gD4, 256, 0, stream>>>(hbf, w1b + w1Off, b1 + l * D4,
                                              nullptr, nullptr, m1bf, BT, D4, Dd, 2);
    // x = x + m1 @ W2 + b2
    gemm_bf16_kernel<<<gD, 256, 0, stream>>>(m1bf, w2b + w1Off, b2 + l * Dd, xf,
                                             xf, nullptr, BT, Dd, D4, 1);
  }

  // Final LN + LM head
  ln_kernel<<<BT, 256, 0, stream>>>(xf, lnfg, lnfb, hbf, Dd);
  gemm_bf16_kernel<<<gV, 256, 0, stream>>>(hbf, wlmb, blm, nullptr, logits,
                                           nullptr, BT, Vv, Dd, 3);
}
